// LSTM_15504831938619
// MI455X (gfx1250) — matrix operand order repaired and checked by reading, not compiled, not run on hardware
//
#include <hip/hip_runtime.h>
#include <hip/hip_bf16.h>

// ---------------------------------------------------------------------------
// Masked LSTM, T=256, B=64, OBS=H=512, fused 4-gate weight [1024, 2048].
//
// MI455X strategy (wave32, WMMA bf16, f32 accum):
//   K0  pack_w:   W -> bf16 B-fragments (Wx/Wh) in documented 32x16 B layout.
//   K1  gemm1:    Z[T*B,4H] = x @ Wx + b.  Per block: stage the shared 16x512
//                 x tile into LDS as bf16 A-fragments once (8 waves share it);
//                 software-pipelined kb loop (prefetch kb+1 while WMMA kb).
//   K2  lstm_rec: persistent kernel, 64 blocks x 64 thr = 128 waves
//                 = 4 b-tiles x 32 j-tiles.  Each block ASYNC-copies its
//                 128KB of loop-invariant Wh B-fragments into LDS ONCE
//                 (global_load_async_to_lds_b128 + s_wait_asynccnt).  Per step:
//                   (prefetch Z-frag accumulators, overlaps A+barrier)
//                   A: mask-reset state, h -> bf16 A-fragments   (grid barrier)
//                   B: acc(g) = Z-frag + hbf @ Wh(LDS), double-buffered kb loop
//                   C: sigmoid/tanh + c/h update in-register     (grid barrier)
// ---------------------------------------------------------------------------

typedef __attribute__((ext_vector_type(16))) __bf16 v16bf;
typedef __attribute__((ext_vector_type(8)))  float  v8f;

#define T_STEPS 256
#define BATCH   64
#define OBS     512
#define HID     512

// workspace byte offsets (all 1KB aligned)
#define WS_BAR    0          // 2 x u32 barrier state (256B reserved)
#define WS_BIAS   1024       // 2048 f32                      (8 KB)
#define WS_WX     16384      // 2048 frags x 1KB bf16         (2 MB)
#define WS_WH     (WS_WX  + 2097152)
#define WS_HBF    (WS_WH  + 2097152)   // 64 frags x 1KB      (64 KB)
#define WS_HST    (WS_HBF + 65536)     // 64x512 f32          (128 KB)
#define WS_CST    (WS_HST + 131072)    // 64x512 f32          (128 KB)
#define WS_Z      (WS_CST + 131072)    // 131072 frags x 1KB  (134 MB)

__device__ __forceinline__ float sig_(float x) {
  return 1.0f / (1.0f + __expf(-x));
}
__device__ __forceinline__ float tanh_(float x) {
  return 1.0f - 2.0f / (__expf(2.0f * x) + 1.0f);   // well-behaved at +/-inf
}

// A 16x32 bf16 layout inverse: (m, kk) -> (lane, elem)
__device__ __forceinline__ void a_frag_pos(int m, int kk, int* lane_a, int* e) {
  if (kk < 8)       { *lane_a = m;      *e = kk;      }
  else if (kk < 16) { *lane_a = m + 16; *e = kk - 8;  }
  else if (kk < 24) { *lane_a = m;      *e = kk - 8;  }
  else              { *lane_a = m + 16; *e = kk - 16; }
}

#define WMMA_BF16(A, B, C) \
  __builtin_amdgcn_wmma_f32_16x16x32_bf16(false, (A), false, (B), (short)0, (C), false, false)

// ---------------------------------------------------------------------------
// K0: pack W (4 gate matrices [1024,512] f32) into bf16 B-fragments.
// B 32x16 bf16 layout (wave32): lane l: N = l&15, K = (l>=16 ? 8:0) + (e<8 ? e : e+8).
// Fragment id = ct*16 + kb, 512 bf16 each.
// ---------------------------------------------------------------------------
__global__ void lstm_pack_w(const float* __restrict__ Wf, const float* __restrict__ bf,
                            const float* __restrict__ Wi, const float* __restrict__ bi,
                            const float* __restrict__ Wo, const float* __restrict__ bo,
                            const float* __restrict__ Wc, const float* __restrict__ bc,
                            __bf16* __restrict__ wx, __bf16* __restrict__ wh,
                            float* __restrict__ bias) {
  int idx = blockIdx.x * 256 + threadIdx.x;       // over 1024*2048
  if (idx >= 1024 * 2048) return;
  int r = idx >> 11;          // row in fused W   [0,1024)
  int c = idx & 2047;         // col in fused W   [0,2048)
  int gate = c >> 9;
  int wc   = c & 511;
  const float* Wsrc = (gate == 0) ? Wf : (gate == 1) ? Wi : (gate == 2) ? Wo : Wc;
  float w = Wsrc[r * 512 + wc];

  int k = (r < 512) ? r : (r - 512);
  __bf16* dst = (r < 512) ? wx : wh;
  int kb = k >> 5, kk = k & 31;
  int ct = c >> 4, n = c & 15;
  int lane = n + (((kk >> 3) & 1) ? 16 : 0);
  int e = (kk & 7) + ((kk >> 4) << 3);
  dst[(((ct << 4) + kb) << 9) + lane * 16 + e] = (__bf16)w;

  if (r == 0) {
    const float* bsrc = (gate == 0) ? bf : (gate == 1) ? bi : (gate == 2) ? bo : bc;
    bias[c] = bsrc[wc];
  }
}

// ---------------------------------------------------------------------------
// K1: Z = x @ Wx + bias.  M=16384, N=2048, K=512.
// Block = 256 thr = 8 waves, all sharing one mt (mt = blockIdx>>2).
// x tile staged once into LDS as bf16 A-fragments (16 frags x 1KB = 16KB).
// kb loop software-pipelined: prefetch kb+1 fragments during kb WMMAs.
// ---------------------------------------------------------------------------
__global__ void lstm_gemm1(const float* __restrict__ x,
                           const __bf16* __restrict__ wx,
                           const float* __restrict__ bias,
                           float* __restrict__ zfrag) {
  __shared__ __align__(32) __bf16 asmem[16 * 512];   // 16 KB

  const int tid  = threadIdx.x;
  const int lane = tid & 31;
  const int mt   = blockIdx.x >> 2;                  // 0..1023
  const int ctg  = (blockIdx.x & 3) * 8 + (tid >> 5);// 0..31, ct = ctg*4+q
  const int ln   = lane & 15;

  // ---- stage x tile [16 x 512] f32 -> bf16 A-fragments in LDS ----
  const float* xt = x + (long)mt * 16 * OBS;
#pragma unroll
  for (int i = 0; i < 8; ++i) {
    const int idx4 = tid + 256 * i;                  // 2048 float4s
    const int m  = idx4 >> 7;
    const int k0 = (idx4 & 127) * 4;
    float4 v = *(const float4*)(xt + m * 512 + k0);
    float vv[4] = {v.x, v.y, v.z, v.w};
#pragma unroll
    for (int u = 0; u < 4; ++u) {
      const int k = k0 + u;
      const int kb = k >> 5, kk = k & 31;
      int lane_a, e;
      a_frag_pos(m, kk, &lane_a, &e);
      asmem[(kb << 9) + lane_a * 16 + e] = (__bf16)vv[u];
    }
  }
  __syncthreads();

  // per-q B fragment base pointers (advance by 512 elems per kb)
  const __bf16* w0 = wx + (((ctg * 4 + 0) << 4) << 9) + lane * 16;
  const __bf16* w1 = wx + (((ctg * 4 + 1) << 4) << 9) + lane * 16;
  const __bf16* w2 = wx + (((ctg * 4 + 2) << 4) << 9) + lane * 16;
  const __bf16* w3 = wx + (((ctg * 4 + 3) << 4) << 9) + lane * 16;
  const __bf16* ab = asmem + lane * 16;

  v8f acc[4] = {v8f{}, v8f{}, v8f{}, v8f{}};

  v16bf a_c  = *(const v16bf*)(ab);
  v16bf b_c0 = *(const v16bf*)(w0);
  v16bf b_c1 = *(const v16bf*)(w1);
  v16bf b_c2 = *(const v16bf*)(w2);
  v16bf b_c3 = *(const v16bf*)(w3);

#pragma unroll
  for (int kb = 0; kb < 16; ++kb) {
    const int kn = (kb + 1) & 15;                    // branchless prefetch idx
    v16bf a_n  = *(const v16bf*)(ab + (kn << 9));
    v16bf b_n0 = *(const v16bf*)(w0 + (kn << 9));
    v16bf b_n1 = *(const v16bf*)(w1 + (kn << 9));
    v16bf b_n2 = *(const v16bf*)(w2 + (kn << 9));
    v16bf b_n3 = *(const v16bf*)(w3 + (kn << 9));
    acc[0] = WMMA_BF16(a_c, b_c0, acc[0]);
    acc[1] = WMMA_BF16(a_c, b_c1, acc[1]);
    acc[2] = WMMA_BF16(a_c, b_c2, acc[2]);
    acc[3] = WMMA_BF16(a_c, b_c3, acc[3]);
    a_c = a_n; b_c0 = b_n0; b_c1 = b_n1; b_c2 = b_n2; b_c3 = b_n3;
  }

#pragma unroll
  for (int q = 0; q < 4; ++q) {
    const int ct = ctg * 4 + q;
    const float bv = bias[ct * 16 + ln];
    v8f r = acc[q];
#pragma unroll
    for (int e = 0; e < 8; ++e) r[e] += bv;
    *(v8f*)(zfrag + ((long)mt * 128 + ct) * 256 + lane * 8) = r;
  }
}

// ---------------------------------------------------------------------------
// grid-wide barrier (sense via generation counter; blocks all resident)
// ---------------------------------------------------------------------------
__device__ __forceinline__ void grid_barrier(unsigned* cnt, unsigned* gen,
                                             unsigned nb) {
  __syncthreads();
  if (threadIdx.x == 0) {
    unsigned g = __hip_atomic_load(gen, __ATOMIC_RELAXED, __HIP_MEMORY_SCOPE_AGENT);
    unsigned a = __hip_atomic_fetch_add(cnt, 1u, __ATOMIC_ACQ_REL, __HIP_MEMORY_SCOPE_AGENT);
    if (a == nb - 1) {
      __hip_atomic_store(cnt, 0u, __ATOMIC_RELAXED, __HIP_MEMORY_SCOPE_AGENT);
      __hip_atomic_fetch_add(gen, 1u, __ATOMIC_RELEASE, __HIP_MEMORY_SCOPE_AGENT);
    } else {
      while (__hip_atomic_load(gen, __ATOMIC_ACQUIRE, __HIP_MEMORY_SCOPE_AGENT) == g) {
        __builtin_amdgcn_s_sleep(1);
      }
    }
  }
  __syncthreads();
}

// ---------------------------------------------------------------------------
// K2: persistent recurrence. 64 blocks x 64 threads = 128 waves
//     wave -> (bt = gwave>>5 in 0..3, jt = gwave&31 in 0..31)
// ---------------------------------------------------------------------------
__global__ void lstm_rec(const int* __restrict__ mask,
                         const __bf16* __restrict__ wh,
                         const float* __restrict__ zfrag,
                         __bf16* __restrict__ hbf,
                         float* __restrict__ hstate,
                         float* __restrict__ cstate,
                         float* __restrict__ out,
                         unsigned* __restrict__ bar) {
  __shared__ __align__(32) __bf16 whs[2 * 4 * 16 * 512];   // 128 KB

  const int tid     = threadIdx.x;                 // 0..63
  const int lane    = tid & 31;
  const int w       = tid >> 5;                    // wave in block: 0..1
  const int gthread = blockIdx.x * 64 + tid;       // 0..4095
  const int gwave   = blockIdx.x * 2 + w;          // 0..127
  const int bt = gwave >> 5;                       // 0..3
  const int jt = gwave & 31;                       // 0..31
  const int hi = lane >> 4;
  const int ln = lane & 15;
  unsigned* cnt = bar;
  unsigned* gen = bar + 1;

  // ---- one-time: async-copy loop-invariant Wh B-fragments into LDS ----
  // per-wave region: 4 gates x 16 kb x 1KB = 64KB; chunk = 16B (b128)
  {
    const unsigned ldsbase = (unsigned)(unsigned long long)(const void*)whs;
    for (int i = 0; i < 128; ++i) {
      const int c    = tid + 64 * i;               // 8192 chunks of 16B
      const int cw   = c >> 12;                    // which wave's region
      const int rem  = c & 4095;
      const int frag = rem >> 6;                   // g*16 + kb
      const int cif  = rem & 63;                   // chunk within 1KB frag
      const int g    = frag >> 4;
      const int kb   = frag & 15;
      const int jtw  = (blockIdx.x * 2 + cw) & 31;
      const long elemoff = ((long)((g * 32 + jtw) * 16 + kb) << 9) + (long)cif * 8;
      const unsigned lds = ldsbase + (unsigned)c * 16u;
      const unsigned long long ga = (unsigned long long)(wh + elemoff);
      asm volatile("global_load_async_to_lds_b128 %0, %1, off"
                   :: "v"(lds), "v"(ga) : "memory");
    }
    asm volatile("s_wait_asynccnt 0x0" ::: "memory");
  }
  __syncthreads();

  const __bf16* abase = hbf + ((bt << 4) << 9) + lane * 16;    // + kb*512
  const __bf16* wbase = whs + w * 32768 + lane * 16;           // + g*8192 + kb*512

  for (int t = 0; t < T_STEPS; ++t) {
    // ---- prefetch Z accumulators (Z is ready before launch; overlaps
    //      Phase A work and the barrier wait) ----
    v8f acc[4];
    const long mt = (long)t * 4 + bt;
#pragma unroll
    for (int g = 0; g < 4; ++g)
      acc[g] = *(const v8f*)(zfrag + (mt * 128 + g * 32 + jt) * 256 + lane * 8);

    // ---- Phase A: mask-reset state, emit bf16 A-fragments of h ----
#pragma unroll
    for (int i = 0; i < 8; ++i) {
      const int idx = gthread + i * 4096;          // 0..32767
      const int b = idx >> 9, j = idx & 511;
      const float keep = 1.0f - (float)mask[t * BATCH + b];
      const float hp = (t == 0) ? 0.0f : hstate[idx];
      const float cp = (t == 0) ? 0.0f : cstate[idx];
      cstate[idx] = keep * cp;
      const int kb = j >> 5, kk = j & 31;
      const int m = b & 15, btl = b >> 4;
      int lane_a, e;
      a_frag_pos(m, kk, &lane_a, &e);
      hbf[(((btl << 4) + kb) << 9) + lane_a * 16 + e] = (__bf16)(keep * hp);
    }
    grid_barrier(cnt, gen, gridDim.x);

    // ---- Phase B: gates = Z + hbf @ Wh(LDS), double-buffered kb loop ----
    v16bf a_c  = *(const v16bf*)(abase);
    v16bf b_c0 = *(const v16bf*)(wbase + 0 * 8192);
    v16bf b_c1 = *(const v16bf*)(wbase + 1 * 8192);
    v16bf b_c2 = *(const v16bf*)(wbase + 2 * 8192);
    v16bf b_c3 = *(const v16bf*)(wbase + 3 * 8192);
#pragma unroll
    for (int kb = 0; kb < 16; ++kb) {
      const int kn = (kb + 1) & 15;                // branchless prefetch idx
      v16bf a_n  = *(const v16bf*)(abase + (kn << 9));
      v16bf b_n0 = *(const v16bf*)(wbase + 0 * 8192 + (kn << 9));
      v16bf b_n1 = *(const v16bf*)(wbase + 1 * 8192 + (kn << 9));
      v16bf b_n2 = *(const v16bf*)(wbase + 2 * 8192 + (kn << 9));
      v16bf b_n3 = *(const v16bf*)(wbase + 3 * 8192 + (kn << 9));
      acc[0] = WMMA_BF16(a_c, b_c0, acc[0]);
      acc[1] = WMMA_BF16(a_c, b_c1, acc[1]);
      acc[2] = WMMA_BF16(a_c, b_c2, acc[2]);
      acc[3] = WMMA_BF16(a_c, b_c3, acc[3]);
      a_c = a_n; b_c0 = b_n0; b_c1 = b_n1; b_c2 = b_n2; b_c3 = b_n3;
    }

    // ---- Phase C: activations + state update, in-register ----
#pragma unroll
    for (int e = 0; e < 8; ++e) {
      const int m = e + hi * 8;               // C/D frag: VGPR e; lanes>=16 -> M+8
      const int b = bt * 16 + m;
      const int j = jt * 16 + ln;
      const int sidx = b * HID + j;
      const float fg = sig_(acc[0][e]);
      const float ig = sig_(acc[1][e]);
      const float og = sig_(acc[2][e]);
      const float gg = tanh_(acc[3][e]);
      const float cn = fg * cstate[sidx] + ig * gg;
      const float hn = og * tanh_(cn);
      cstate[sidx] = cn;
      hstate[sidx] = hn;
      out[((long)t * BATCH + b) * HID + j] = hn;
      if (t == T_STEPS - 1) {
        out[(long)T_STEPS * BATCH * HID + sidx] = hn;                 // final h
        out[(long)T_STEPS * BATCH * HID + BATCH * HID + sidx] = cn;   // final c
      }
    }
    grid_barrier(cnt, gen, gridDim.x);
  }
}

// ---------------------------------------------------------------------------
extern "C" void kernel_launch(void* const* d_in, const int* in_sizes, int n_in,
                              void* d_out, int out_size, void* d_ws, size_t ws_size,
                              hipStream_t stream) {
  const float* x    = (const float*)d_in[0];
  const int*   mask = (const int*)d_in[1];
  const float* Wf   = (const float*)d_in[2];
  const float* bf   = (const float*)d_in[3];
  const float* Wi   = (const float*)d_in[4];
  const float* bi   = (const float*)d_in[5];
  const float* Wo   = (const float*)d_in[6];
  const float* bo   = (const float*)d_in[7];
  const float* Wc   = (const float*)d_in[8];
  const float* bc   = (const float*)d_in[9];
  float* out = (float*)d_out;

  char* ws = (char*)d_ws;
  unsigned* bar   = (unsigned*)(ws + WS_BAR);
  float*   bias   = (float*)(ws + WS_BIAS);
  __bf16*  wx     = (__bf16*)(ws + WS_WX);
  __bf16*  wh     = (__bf16*)(ws + WS_WH);
  __bf16*  hbf    = (__bf16*)(ws + WS_HBF);
  float*   hstate = (float*)(ws + WS_HST);
  float*   cstate = (float*)(ws + WS_CST);
  float*   zfrag  = (float*)(ws + WS_Z);

  // reset barrier counters every call (graph-capture safe)
  hipMemsetAsync(bar, 0, 256, stream);

  // K0: pack weights (1024*2048 elements)
  lstm_pack_w<<<(1024 * 2048) / 256, 256, 0, stream>>>(Wf, bf, Wi, bi, Wo, bo,
                                                       Wc, bc, wx, wh, bias);
  // K1: big parallel GEMM (8 waves/block share one A tile via LDS)
  lstm_gemm1<<<4096, 256, 0, stream>>>(x, wx, bias, zfrag);

  // K2: persistent recurrence (64 blocks x 64 threads = 128 waves)
  lstm_rec<<<64, 64, 0, stream>>>(mask, wh, zfrag, hbf, hstate, cstate, out, bar);
}
